// MoELayer_65807488910123
// MI455X (gfx1250) — matrix operand order repaired and checked by reading, not compiled, not run on hardware
//
#include <hip/hip_runtime.h>
#include <hip/hip_bf16.h>
#include <type_traits>

// ---------------------------------------------------------------------------
// MoE layer for MI455X (gfx1250): bf16 WMMA GEMMs with fp32 accumulate.
// Compute-bound (~1.24 TFLOP vs ~15us HBM traffic) -> keep the WMMA pipe fed:
//  - weights/x pre-converted (weights pre-transposed into the WMMA-B
//    packed-pair layout) so the GEMM hot loop does straight b128 copies,
//  - GLOBAL_LOAD_ASYNC_TO_LDS_B128 (ASYNCcnt) for the tile staging when the
//    toolchain exposes it, register-staged copies otherwise,
//  - v_cvt_pk_bf16_f32 for remaining conversions when available.
// ---------------------------------------------------------------------------

#define DEV __device__ __forceinline__

typedef __attribute__((ext_vector_type(16))) __bf16          bf16x16;
typedef __attribute__((ext_vector_type(8)))  float           f32x8;
typedef __attribute__((ext_vector_type(16))) unsigned short  u16x16;
typedef __attribute__((ext_vector_type(8)))  unsigned short  u16x8;

static constexpr int N_TOK = 8192;   // tokens
static constexpr int DIM   = 1024;   // input dim
static constexpr int NEXP  = 8;      // experts
static constexpr int HID   = 4096;   // expert hidden (= gate hidden 4*DIM)
static constexpr int OUTD  = 1024;   // expert out

static constexpr int BM = 128, BN = 128, BK = 32, TPB = 256;

#if defined(__HIP_DEVICE_COMPILE__) && \
    __has_builtin(__builtin_amdgcn_global_load_async_to_lds_b128)
#define HAVE_ASYNC_LDS 1
#else
#define HAVE_ASYNC_LDS 0
#endif

DEV unsigned short f2bf(float f) {            // RNE float -> bf16 bits
  unsigned int u = __builtin_bit_cast(unsigned int, f);
  u += 0x7FFFu + ((u >> 16) & 1u);
  return (unsigned short)(u >> 16);
}
DEV float bf2f(unsigned short s) {
  unsigned int u = ((unsigned int)s) << 16;
  return __builtin_bit_cast(float, u);
}
// packed pair {lo, hi<<16}; single VALU op when the target has the cvt
DEV unsigned int cvt_pk_bf16(float lo, float hi) {
#if __has_builtin(__builtin_amdgcn_cvt_pk_bf16_f32)
  auto v = __builtin_amdgcn_cvt_pk_bf16_f32(lo, hi);
  return __builtin_bit_cast(unsigned int, v);
#else
  return ((unsigned int)f2bf(hi) << 16) | (unsigned int)f2bf(lo);
#endif
}

#if HAVE_ASYNC_LDS
// builtin prototype (from hipcc diagnostic): params are vector_size(16) int
// pointers; AS(1) global source, AS(3) LDS destination, imm offset, imm cpol.
typedef int v4i_vs __attribute__((vector_size(16)));
typedef __attribute__((address_space(1))) v4i_vs g_v4i;
typedef __attribute__((address_space(3))) v4i_vs l_v4i;

DEV void async_cp16(const void* g, void* l) {   // 16B global -> LDS, ASYNCcnt
  __builtin_amdgcn_global_load_async_to_lds_b128((g_v4i*)g, (l_v4i*)l, 0, 0);
}
DEV void wait_async0() {
#if __has_builtin(__builtin_amdgcn_s_wait_asynccnt)
  __builtin_amdgcn_s_wait_asynccnt(0);
#else
  asm volatile("s_wait_asynccnt 0" ::: "memory");
#endif
}
#endif

union F8R { float4 v[2]; float f[8]; };

enum { EPI_RELU_BF16 = 0, EPI_COMBINE_F32 = 1 };

// ---------------------------------------------------------------------------
// fp32 -> packed bf16, 8 elements per thread.
// ---------------------------------------------------------------------------
__global__ __launch_bounds__(256)
void convert_f32_bf16(const float* __restrict__ src, unsigned int* __restrict__ dst) {
  const size_t g = (size_t)blockIdx.x * 256 + threadIdx.x;
  const float4 a = *(const float4*)(src + g * 8);
  const float4 b = *(const float4*)(src + g * 8 + 4);
  uint4 o;
  o.x = cvt_pk_bf16(a.x, a.y);
  o.y = cvt_pk_bf16(a.z, a.w);
  o.z = cvt_pk_bf16(b.x, b.y);
  o.w = cvt_pk_bf16(b.z, b.w);
  *(uint4*)(dst + g * 4) = o;
}

// ---------------------------------------------------------------------------
// Weight pre-pass: W[K][Ncols] fp32 -> Bp[Ncols][K/2] packed bf16 K-pairs
// (exactly the dwords the WMMA-B LDS tile wants). LDS-tiled so both global
// read and write stay coalesced. grid = (Ncols/64, K/128, numMatrices).
// ---------------------------------------------------------------------------
__global__ __launch_bounds__(256)
void transpose_pack_bf16(const float* __restrict__ W, unsigned int* __restrict__ Bp,
                         int K, int Ncols) {
  __shared__ float sT[128][65];     // +1 pad: conflict-free column reads
  const int t  = threadIdx.x;
  const int n0 = blockIdx.x * 64;
  const int k0 = blockIdx.y * 128;
  const size_t matOff = (size_t)blockIdx.z * K * Ncols;
  const float* Wm = W + matOff;
  unsigned int* Bm = Bp + matOff / 2;

#pragma unroll
  for (int i = 0; i < 32; ++i) {
    const int idx = t + i * 256;            // 0..8191
    const int k = idx >> 6, n = idx & 63;
    sT[k][n] = Wm[(size_t)(k0 + k) * Ncols + n0 + n];
  }
  __syncthreads();
#pragma unroll
  for (int i = 0; i < 16; ++i) {
    const int idx = t + i * 256;            // 0..4095
    const int n = idx >> 6, kp = idx & 63;
    Bm[(size_t)(n0 + n) * (K / 2) + (k0 >> 1) + kp] =
        cvt_pk_bf16(sT[2 * kp][n], sT[2 * kp + 1][n]);
  }
}

// ---------------------------------------------------------------------------
// Tiled GEMM: C[M,Ncols] = epilogue( A[M,K](bf16) @ B[K,Ncols] + bias )
//  BPACKED=1: B already in packed-pair [Ncols][K/2] layout (pure copies;
//             async global->LDS staging when available).
//  BPACKED=0: B fp32 row-major, packed on the fly with cvt_pk.
//  8 waves, wave (wM,wN) in 4x2 grid owns 32x64 = 2x4 WMMA tiles.
// ---------------------------------------------------------------------------
template <bool BPACKED, int EPI>
__global__ __launch_bounds__(TPB)
void gemm_bf16_wmma(const unsigned short* __restrict__ A, const void* __restrict__ Bv,
                    const float* __restrict__ bias,
                    unsigned short* __restrict__ Cb,       // EPI_RELU_BF16 dst
                    float* __restrict__ Cf,                // EPI_COMBINE dst
                    const float* __restrict__ scale,       // p + e (stride NEXP)
                    int scaleStride, int accumulate,
                    int K, int Ncols) {
  __shared__ alignas(64) unsigned short sA[2][BM][BK];
  __shared__ alignas(64) unsigned int   sB[2][BN][BK / 2];

  const int t    = threadIdx.x;
  const int bm   = blockIdx.y, bn = blockIdx.x;
  const int wid  = t >> 5, lane = t & 31;
  const int wM   = wid & 3;              // 4 waves along M (32 rows each)
  const int wN   = wid >> 2;             // 2 waves along N (64 cols each)
  const int lr   = lane & 15, lh = lane >> 4;

  // ---- A: 2 threads/row, 16 bf16 (32B) each, straight copy ----
  const int ar = t >> 1;
  const int ak = (t & 1) * 16;
  const unsigned short* Ag = A + (size_t)(bm * BM + ar) * K + ak;

  // ---- B mappings ----
  // packed: 2 threads/col, 8 dwords (32B) each, straight copy
  const int nl = t >> 1, bh = t & 1;
  const unsigned int* Bgp =
      BPACKED ? (const unsigned int*)Bv + (size_t)(bn * BN + nl) * (K / 2) + bh * 8
              : nullptr;
  // fp32: K-pair 0..15 x 8-col group, cvt_pk on the fly
  const int bkp = t >> 4;
  const int bc  = (t & 15) * 8;
  const float* Bgf =
      BPACKED ? nullptr
              : (const float*)Bv + (size_t)(2 * bkp) * Ncols + (size_t)bn * BN + bc;

  uint4 aU0, aU1, bU0, bU1;
  F8R bR0, bR1;

  auto gload = [&](int kt) {
    const uint4* pa = (const uint4*)(Ag + (size_t)kt * BK);
    aU0 = pa[0]; aU1 = pa[1];
    if constexpr (BPACKED) {
      const uint4* pb = (const uint4*)(Bgp + (size_t)kt * (BK / 2));
      bU0 = pb[0]; bU1 = pb[1];
    } else {
      const float* pb = Bgf + (size_t)kt * BK * Ncols;
      bR0.v[0] = *(const float4*)(pb);
      bR0.v[1] = *(const float4*)(pb + 4);
      bR1.v[0] = *(const float4*)(pb + Ncols);
      bR1.v[1] = *(const float4*)(pb + Ncols + 4);
    }
  };

  auto sstore = [&](int buf) {
    *(uint4*)&sA[buf][ar][ak]     = aU0;
    *(uint4*)&sA[buf][ar][ak + 8] = aU1;
    if constexpr (BPACKED) {
      *(uint4*)&sB[buf][nl][bh * 8]     = bU0;
      *(uint4*)&sB[buf][nl][bh * 8 + 4] = bU1;
    } else {
#pragma unroll
      for (int i = 0; i < 8; ++i)
        sB[buf][bc + i][bkp] = cvt_pk_bf16(bR0.f[i], bR1.f[i]);
    }
  };

  f32x8 acc[2][4];
#pragma unroll
  for (int m = 0; m < 2; ++m)
#pragma unroll
    for (int n = 0; n < 4; ++n) acc[m][n] = (f32x8)(0.0f);

  auto compute = [&](int buf) {
    bf16x16 af[2], bfg[4];
#pragma unroll
    for (int m = 0; m < 2; ++m) {
      // contiguous K order (matches B): lanes 0-15 K{0..15}, 16-31 K{16..31}
      const unsigned short* pa = &sA[buf][wM * 32 + m * 16 + lr][lh * 16];
      u16x8 lo = *(const u16x8*)pa;
      u16x8 hi = *(const u16x8*)(pa + 8);
      u16x16 cat = __builtin_shufflevector(lo, hi, 0, 1, 2, 3, 4, 5, 6, 7,
                                           8, 9, 10, 11, 12, 13, 14, 15);
      af[m] = __builtin_bit_cast(bf16x16, cat);
    }
#pragma unroll
    for (int n = 0; n < 4; ++n) {
      const unsigned int* pb = &sB[buf][wN * 64 + n * 16 + lr][lh * 8];
      u16x16 bb = *(const u16x16*)pb;        // two ds_load_b128
      bfg[n] = __builtin_bit_cast(bf16x16, bb);
    }
#pragma unroll
    for (int m = 0; m < 2; ++m)
#pragma unroll
      for (int n = 0; n < 4; ++n)
        acc[m][n] = __builtin_amdgcn_wmma_f32_16x16x32_bf16(
            false, af[m], false, bfg[n], (short)0, acc[m][n], false, false);
  };

  const int KT = K / BK;
#if HAVE_ASYNC_LDS
  if constexpr (BPACKED) {
    // DMA-style staging: global -> LDS without VGPR round trip (ASYNCcnt)
    auto stageAsync = [&](int kt, int buf) {
      async_cp16(Ag + (size_t)kt * BK,      &sA[buf][ar][ak]);
      async_cp16(Ag + (size_t)kt * BK + 8,  &sA[buf][ar][ak + 8]);
      async_cp16(Bgp + (size_t)kt * (BK / 2),     &sB[buf][nl][bh * 8]);
      async_cp16(Bgp + (size_t)kt * (BK / 2) + 4, &sB[buf][nl][bh * 8 + 4]);
    };
    stageAsync(0, 0);
    wait_async0();
    __syncthreads();
    for (int kt = 0; kt < KT; ++kt) {
      const int cur = kt & 1;
      if (kt + 1 < KT) stageAsync(kt + 1, cur ^ 1);   // DMA overlaps WMMAs
      compute(cur);
      if (kt + 1 < KT) {
        wait_async0();
        __syncthreads();
      }
    }
  } else
#endif
  {
    gload(0);
    sstore(0);
    __syncthreads();
    for (int kt = 0; kt < KT; ++kt) {
      const int cur = kt & 1;
      if (kt + 1 < KT) gload(kt + 1);
      if (kt + 2 < KT) {                         // global_prefetch_b8
        __builtin_prefetch((const void*)(Ag + (size_t)(kt + 2) * BK), 0, 1);
        if constexpr (BPACKED)
          __builtin_prefetch((const void*)(Bgp + (size_t)(kt + 2) * (BK / 2)), 0, 1);
        else
          __builtin_prefetch((const void*)(Bgf + (size_t)(kt + 2) * BK * Ncols), 0, 1);
      }
      compute(cur);
      if (kt + 1 < KT) {
        sstore(cur ^ 1);
        __syncthreads();
      }
    }
  }

  // ---- epilogue (C/D layout: VGPR v -> row base+v+8*lh, col base+lr) ----
  // 32-bit indexing: all matrices < 2^31 elements -> single v_mul_lo_u32.
  const int rowBase = bm * BM + wM * 32;
  const int colBase = bn * BN + wN * 64;
#pragma unroll
  for (int m = 0; m < 2; ++m) {
#pragma unroll
    for (int n = 0; n < 4; ++n) {
      const unsigned col = (unsigned)(colBase + n * 16 + lr);
      const float bv = bias[col];
      const int r0 = rowBase + m * 16 + lh * 8;
#pragma unroll
      for (int v = 0; v < 8; ++v) {
        const unsigned row = (unsigned)(r0 + v);
        const unsigned idx = row * (unsigned)Ncols + col;
        float val = acc[m][n][v] + bv;
        if constexpr (EPI == EPI_RELU_BF16) {
          val = fmaxf(val, 0.0f);
          Cb[idx] = f2bf(val);
        } else {
          const float s = scale[row * (unsigned)scaleStride];
          const float contrib = s * val;
          if (accumulate) Cf[idx] += contrib; else Cf[idx] = contrib;
        }
      }
    }
  }
}

// ---------------------------------------------------------------------------
// Gate logits (h2[N,D] bf16 @ G3[D,E] + g3), top-2 softmax -> dense p[N,E].
// ---------------------------------------------------------------------------
__global__ __launch_bounds__(256)
void gate_logits_topk(const unsigned short* __restrict__ h2,
                      const float* __restrict__ G3, const float* __restrict__ g3,
                      float* __restrict__ p) {
  const int n = blockIdx.x, t = threadIdx.x;
  float part[NEXP];
#pragma unroll
  for (int e = 0; e < NEXP; ++e) part[e] = 0.0f;

  const unsigned short* hrow = h2 + (size_t)n * DIM;
  for (int k = t; k < DIM; k += 256) {
    const float hv = bf2f(hrow[k]);
    const float* g = G3 + (size_t)k * NEXP;
#pragma unroll
    for (int e = 0; e < NEXP; ++e) part[e] += hv * g[e];
  }

  __shared__ float red[256][NEXP];
#pragma unroll
  for (int e = 0; e < NEXP; ++e) red[t][e] = part[e];
  __syncthreads();
  for (int s = 128; s > 0; s >>= 1) {
    if (t < s) {
#pragma unroll
      for (int e = 0; e < NEXP; ++e) red[t][e] += red[t + s][e];
    }
    __syncthreads();
  }

  if (t == 0) {
    float lg[NEXP];
#pragma unroll
    for (int e = 0; e < NEXP; ++e) lg[e] = red[0][e] + g3[e];
    int i0 = 0;
    for (int e = 1; e < NEXP; ++e) if (lg[e] > lg[i0]) i0 = e;
    int i1 = -1;
    for (int e = 0; e < NEXP; ++e)
      if (e != i0 && (i1 < 0 || lg[e] > lg[i1])) i1 = e;
    const float e1 = __expf(lg[i1] - lg[i0]);
    const float inv = 1.0f / (1.0f + e1);
    float out[NEXP];
#pragma unroll
    for (int e = 0; e < NEXP; ++e) out[e] = 0.0f;
    out[i0] = inv;
    out[i1] = e1 * inv;
#pragma unroll
    for (int e = 0; e < NEXP; ++e) p[(size_t)n * NEXP + e] = out[e];
  }
}

// ---------------------------------------------------------------------------
extern "C" void kernel_launch(void* const* d_in, const int* in_sizes, int n_in,
                              void* d_out, int out_size, void* d_ws,
                              size_t ws_size, hipStream_t stream) {
  (void)in_sizes; (void)n_in; (void)out_size;

  const float* x  = (const float*)d_in[0];   // [N, D]
  const float* W1 = (const float*)d_in[1];   // [E, D, H]
  const float* b1 = (const float*)d_in[2];   // [E, H]
  const float* W2 = (const float*)d_in[3];   // [E, H, O]
  const float* b2 = (const float*)d_in[4];   // [E, O]
  const float* G1 = (const float*)d_in[5];   // [D, 4D]
  const float* g1 = (const float*)d_in[6];
  const float* G2 = (const float*)d_in[7];   // [4D, D]
  const float* g2 = (const float*)d_in[8];
  const float* G3 = (const float*)d_in[9];   // [D, E]
  const float* g3 = (const float*)d_in[10];

  float* y = (float*)d_out;                        // [N, O]
  float* p = (float*)d_out + (size_t)N_TOK * OUTD; // [N, E]

  // ---- workspace carve ----
  const size_t nXB = (size_t)N_TOK * DIM;          // bf16 elems
  const size_t nH1 = (size_t)N_TOK * HID;
  const size_t nH2 = (size_t)N_TOK * DIM;
  const size_t nG1 = (size_t)DIM * HID / 2;        // packed dwords
  const size_t nG2 = (size_t)HID * OUTD / 2;
  const size_t nW1 = (size_t)NEXP * DIM * HID / 2;
  const size_t nW2 = (size_t)NEXP * HID * OUTD / 2;
  const size_t baseBytes = (nXB + nH1 + nH2) * 2;
  const size_t fullBytes = baseBytes + (nG1 + nG2 + nW1 + nW2) * 4;

  unsigned short* xb = (unsigned short*)d_ws;
  unsigned short* h1 = xb + nXB;                   // gate hidden, then expert hid
  unsigned short* h2 = h1 + nH1;
  unsigned int* G1p = (unsigned int*)(h2 + nH2);
  unsigned int* G2p = G1p + nG1;
  unsigned int* W1p = G2p + nG2;
  unsigned int* W2p = W1p + nW1;
  const bool packed = ws_size >= fullBytes;

  const dim3 blk(TPB);
  const dim3 gridH(HID / BN, N_TOK / BM);    // Ncols = 4096
  const dim3 gridD(OUTD / BN, N_TOK / BM);   // Ncols = 1024

  // x -> bf16 (both paths)
  convert_f32_bf16<<<(N_TOK * DIM) / (8 * 256), 256, 0, stream>>>(
      x, (unsigned int*)xb);

  if (packed) {
    // weight pre-pass into WMMA-B packed layout
    transpose_pack_bf16<<<dim3(HID / 64, DIM / 128, 1), 256, 0, stream>>>(G1, G1p, DIM, HID);
    transpose_pack_bf16<<<dim3(OUTD / 64, HID / 128, 1), 256, 0, stream>>>(G2, G2p, HID, OUTD);
    transpose_pack_bf16<<<dim3(HID / 64, DIM / 128, NEXP), 256, 0, stream>>>(W1, W1p, DIM, HID);
    transpose_pack_bf16<<<dim3(OUTD / 64, HID / 128, NEXP), 256, 0, stream>>>(W2, W2p, HID, OUTD);

    gemm_bf16_wmma<true, EPI_RELU_BF16><<<gridH, blk, 0, stream>>>(
        xb, G1p, g1, h1, nullptr, nullptr, 0, 0, DIM, HID);
    gemm_bf16_wmma<true, EPI_RELU_BF16><<<gridD, blk, 0, stream>>>(
        h1, G2p, g2, h2, nullptr, nullptr, 0, 0, HID, OUTD);
    gate_logits_topk<<<N_TOK, 256, 0, stream>>>(h2, G3, g3, p);
    for (int e = 0; e < NEXP; ++e) {
      gemm_bf16_wmma<true, EPI_RELU_BF16><<<gridH, blk, 0, stream>>>(
          xb, W1p + (size_t)e * DIM * HID / 2, b1 + (size_t)e * HID,
          h1, nullptr, nullptr, 0, 0, DIM, HID);
      gemm_bf16_wmma<true, EPI_COMBINE_F32><<<gridD, blk, 0, stream>>>(
          h1, W2p + (size_t)e * HID * OUTD / 2, b2 + (size_t)e * OUTD,
          nullptr, y, p + e, NEXP, (e > 0) ? 1 : 0, HID, OUTD);
    }
  } else {
    // fallback: B stays fp32, packed on the fly with cvt_pk
    gemm_bf16_wmma<false, EPI_RELU_BF16><<<gridH, blk, 0, stream>>>(
        xb, G1, g1, h1, nullptr, nullptr, 0, 0, DIM, HID);
    gemm_bf16_wmma<false, EPI_RELU_BF16><<<gridD, blk, 0, stream>>>(
        h1, G2, g2, h2, nullptr, nullptr, 0, 0, HID, OUTD);
    gate_logits_topk<<<N_TOK, 256, 0, stream>>>(h2, G3, g3, p);
    for (int e = 0; e < NEXP; ++e) {
      gemm_bf16_wmma<false, EPI_RELU_BF16><<<gridH, blk, 0, stream>>>(
          xb, W1 + (size_t)e * DIM * HID, b1 + (size_t)e * HID,
          h1, nullptr, nullptr, 0, 0, DIM, HID);
      gemm_bf16_wmma<false, EPI_COMBINE_F32><<<gridD, blk, 0, stream>>>(
          h1, W2 + (size_t)e * HID * OUTD, b2 + (size_t)e * OUTD,
          nullptr, y, p + e, NEXP, (e > 0) ? 1 : 0, HID, OUTD);
    }
  }
}
